// YvStableMoEGate_83597243449509
// MI455X (gfx1250) — matrix operand order repaired and checked by reading, not compiled, not run on hardware
//
#include <hip/hip_runtime.h>
#include <hip/hip_bf16.h>

typedef __attribute__((ext_vector_type(16))) __bf16 v16bf;
typedef __attribute__((ext_vector_type(8)))  float  v8f;
typedef __attribute__((ext_vector_type(4)))  unsigned int u32x4;
typedef __attribute__((ext_vector_type(8)))  int i32x8;
typedef __attribute__((ext_vector_type(4)))  int i32x4;

#define H_DIM 4096
#define E_EXP 64
#define ROWS_PER_BLK 128
#define TOPK 2
#define KCHUNKS (H_DIM / 32)
#define CHUNK_ELEMS (128 * 32)   // bf16 elements per staged weight slice

__device__ __forceinline__ float sigmoidf_(float x) { return 1.f / (1.f + __expf(-x)); }

// ---------------------------------------------------------------------------
// Kernel 1: zero the d_ws accumulators (P sums, f counts, comp sum)
// d_ws float layout: [0:64] future_load, [64:128] P, [128:192] cnt, [192] comp
// bf16 weight copy (TDM path) lives at byte offset 4096.
// ---------------------------------------------------------------------------
__global__ void moe_zero_ws(float* __restrict__ ws) {
  const int t = threadIdx.x;
  if (t < 129) ws[64 + t] = 0.f;
}

// ---------------------------------------------------------------------------
// Kernel 1b (TDM path): convert W = [gate_w ; cp_w1] (f32) to bf16, stored
// chunk-major: wbf[c*4096 + row*32 + j] = W[row][c*32 + j]. Each K-chunk is
// then one contiguous 8 KB block -> a single 1-D TDM tile.
// ---------------------------------------------------------------------------
__global__ __launch_bounds__(256) void moe_convert_w(
    const float* __restrict__ gw, const float* __restrict__ w1,
    __bf16* __restrict__ wbf)
{
  const size_t l = ((size_t)blockIdx.x * 256 + threadIdx.x) * 4;   // 4 elems/thread
  const int c   = (int)(l >> 12);
  const int rem = (int)(l & 4095);
  const int row = rem >> 5;
  const int j   = rem & 31;
  const float* src = (row < 64) ? (gw + (size_t)row * H_DIM + c * 32 + j)
                                : (w1 + (size_t)(row - 64) * H_DIM + c * 32 + j);
  float4 v = *(const float4*)src;
  __bf16* d = wbf + l;
  d[0] = (__bf16)v.x; d[1] = (__bf16)v.y; d[2] = (__bf16)v.z; d[3] = (__bf16)v.w;
}

// ---------------------------------------------------------------------------
// TDM helper: one 1-D tensor tile (4096 bf16 = 8 KB) global -> LDS.
// D# group0: count=1 | lds_addr | global_addr | type=2 (ISA 8.3)
// D# group1: data_size=2B, tensor_dim0=4096, tensor_dim1=1, tile_dim0=4096,
//            tile_dim1=1, dim0_stride=4096 (ISA 8.4). Groups 2/3 = 0 (<=2D).
// 6-arg builtin (clang-23 / therock-10.0 flavor).
// ---------------------------------------------------------------------------
__device__ __forceinline__ void tdm_load_chunk(const __bf16* gsrc, unsigned lds_addr) {
  const unsigned long long ga = (unsigned long long)(size_t)gsrc;
  u32x4 g0;
  g0[0] = 1u;                                            // count=1, no gather
  g0[1] = lds_addr;                                      // LDS byte address
  g0[2] = (unsigned)(ga & 0xFFFFFFFFu);                  // global_addr[31:0]
  g0[3] = (unsigned)((ga >> 32) & 0x01FFFFFFu) | (2u << 30); // addr[56:32]|type=2
  i32x8 g1;
  g1[0] = 0x00010000;              // wg_mask=0, data_size=1 (2 bytes)
  g1[1] = (int)(4096u << 16);      // tensor_dim0[15:0]=4096 in bits[63:48]
  g1[2] = 0x00010000;              // tensor_dim0[31:16]=0, tensor_dim1=1
  g1[3] = (int)(4096u << 16);      // tile_dim0=4096 in bits[127:112]
  g1[4] = 1;                       // tile_dim1=1, tile_dim2=0
  g1[5] = 4096;                    // tensor_dim0_stride[31:0]
  g1[6] = 0;
  g1[7] = 0;
  const i32x4 z4 = {0, 0, 0, 0};
  const i32x8 z8 = {0, 0, 0, 0, 0, 0, 0, 0};
  __builtin_amdgcn_tensor_load_to_lds(g0, g1, z4, z4, z8, 0);
}

// ---------------------------------------------------------------------------
// Kernel 2: two-layer LSTM load predictor + head + softmax (tiny, serial).
// ---------------------------------------------------------------------------
__global__ __launch_bounds__(256) void moe_lstm_pred(
    const float* __restrict__ lb,
    const float* __restrict__ wih0, const float* __restrict__ whh0,
    const float* __restrict__ bih0, const float* __restrict__ bhh0,
    const float* __restrict__ wih1, const float* __restrict__ whh1,
    const float* __restrict__ bih1, const float* __restrict__ bhh1,
    const float* __restrict__ head_w, const float* __restrict__ head_b,
    float* __restrict__ fl_ws, float* __restrict__ fl_out)
{
  __shared__ float h[64], c[64], zb[256], h1s[50 * 64], pbuf[640], marr[64];
  const int j = threadIdx.x;

  if (j < 64) { h[j] = 0.f; c[j] = 0.f; }
  __syncthreads();
  for (int t = 0; t < 50; ++t) {
    float z = bih0[j] + bhh0[j];
    const float* wi  = wih0 + j * 64;
    const float* wh  = whh0 + j * 64;
    const float* xin = lb + t * 64;
    #pragma unroll 8
    for (int k = 0; k < 64; ++k) z += wi[k] * xin[k] + wh[k] * h[k];
    zb[j] = z;
    __syncthreads();
    if (j < 64) {
      float ig = sigmoidf_(zb[j]);
      float fg = sigmoidf_(zb[64 + j]);
      float gg = tanhf(zb[128 + j]);
      float og = sigmoidf_(zb[192 + j]);
      float cn = fg * c[j] + ig * gg;
      c[j] = cn;
      float hn = og * tanhf(cn);
      h[j] = hn;
      h1s[t * 64 + j] = hn;
    }
    __syncthreads();
  }
  if (j < 64) { h[j] = 0.f; c[j] = 0.f; }
  __syncthreads();
  for (int t = 0; t < 50; ++t) {
    float z = bih1[j] + bhh1[j];
    const float* wi  = wih1 + j * 64;
    const float* wh  = whh1 + j * 64;
    const float* xin = h1s + t * 64;
    #pragma unroll 8
    for (int k = 0; k < 64; ++k) z += wi[k] * xin[k] + wh[k] * h[k];
    zb[j] = z;
    __syncthreads();
    if (j < 64) {
      float ig = sigmoidf_(zb[j]);
      float fg = sigmoidf_(zb[64 + j]);
      float gg = tanhf(zb[128 + j]);
      float og = sigmoidf_(zb[192 + j]);
      float cn = fg * c[j] + ig * gg;
      c[j] = cn;
      h[j] = og * tanhf(cn);
    }
    __syncthreads();
  }
  for (int jj = j; jj < 640; jj += 256) {
    float z = head_b[jj];
    const float* wr = head_w + jj * 64;
    #pragma unroll 8
    for (int k = 0; k < 64; ++k) z += wr[k] * h[k];
    pbuf[jj] = z;
  }
  __syncthreads();
  if (j < 64) {
    float m = 0.f;
    #pragma unroll
    for (int d = 0; d < 10; ++d) m += pbuf[j * 10 + d];
    marr[j] = m * 0.1f;
  }
  __syncthreads();
  if (j < 64) {
    float mx = -3.0e38f;
    for (int e = 0; e < 64; ++e) mx = fmaxf(mx, marr[e]);
    float s = 0.f;
    for (int e = 0; e < 64; ++e) s += __expf(marr[e] - mx);
    float v = __expf(marr[j] - mx) / s;
    fl_ws[j]  = v;
    fl_out[j] = v;
  }
}

// ---------------------------------------------------------------------------
// Kernel 3: fused gate GEMM + complexity GEMM (bf16 WMMA) + softmax/top-2.
// USE_TDM=1: weights pre-converted to bf16 in d_ws; double-buffered 8 KB
//            slices DMA'd to LDS by the Tensor Data Mover, overlapped with
//            WMMA of the previous chunk (TENSORcnt pipelining).
// USE_TDM=0: synchronous f32->bf16 staging (round-1 path, ws too small).
// ---------------------------------------------------------------------------
template <bool USE_TDM>
__global__ __launch_bounds__(256) void moe_gate_main(
    const float* __restrict__ x,
    const float* __restrict__ gate_w,
    const float* __restrict__ expert_bias,
    const float* __restrict__ cp_w1,
    const float* __restrict__ cp_b1,
    const float* __restrict__ cp_w2,
    const float* __restrict__ cp_b2,
    const __bf16* __restrict__ wbf,         // TDM path only
    const float* __restrict__ future_load,
    float* __restrict__ Pg,
    float* __restrict__ cntg,
    float* __restrict__ compg,
    float* __restrict__ out_scores,
    float* __restrict__ out_idx,
    int N)
{
  __shared__ __bf16 s_stage[(USE_TDM ? 2 : 1) * CHUNK_ELEMS];
  __shared__ float  s_out[8 * 16 * 64];
  __shared__ float  s_eb[64], s_fl[64], s_b1[64], s_w2[64];
  __shared__ float  s_P[64], s_cnt[64], s_comp;

  const int tid  = threadIdx.x;
  const int wave = tid >> 5;
  const int lane = tid & 31;
  const int nloc = lane & 15;
  const int half = lane >> 4;
  const int mblk = blockIdx.x * ROWS_PER_BLK;

  if (tid < 64) {
    s_eb[tid]  = expert_bias[tid];
    s_fl[tid]  = future_load[tid];
    s_b1[tid]  = cp_b1[tid];
    s_w2[tid]  = cp_w2[tid];
    s_P[tid]   = 0.f;
    s_cnt[tid] = 0.f;
  }
  if (tid == 0) s_comp = 0.f;

  const int    wr_row = tid >> 1;
  const int    wr_col = (tid & 1) * 16;
  const float* wsrc   = (wr_row < 64) ? (gate_w + (size_t)wr_row * H_DIM)
                                      : (cp_w1 + (size_t)(wr_row - 64) * H_DIM);

  const float* xrow = x + (size_t)(mblk + wave * 16 + nloc) * H_DIM;
  const unsigned lds_base = (unsigned)(size_t)(void*)&s_stage[0]; // LDS addr = low 32 bits

  v8f acc[8] = {};

  if (USE_TDM) {
    if (wave == 0) tdm_load_chunk(wbf, lds_base);   // prefetch chunk 0
  }

  for (int c = 0; c < KCHUNKS; ++c) {
    const int k0 = c * 32;
    if (USE_TDM) {
      if (wave == 0) {
        if (c + 1 < KCHUNKS) {
          tdm_load_chunk(wbf + (size_t)(c + 1) * CHUNK_ELEMS,
                         lds_base + (unsigned)(((c + 1) & 1) * CHUNK_ELEMS * 2));
          __builtin_amdgcn_s_wait_tensorcnt(1);   // chunk c landed, c+1 in flight
        } else {
          __builtin_amdgcn_s_wait_tensorcnt(0);
        }
      }
      __syncthreads();                            // slice visible to all waves
    } else {
      __syncthreads();
      {
        const float4* p = (const float4*)(wsrc + k0 + wr_col);
        float4 a0 = p[0], a1 = p[1], a2 = p[2], a3 = p[3];
        v16bf wf;
        wf[0]  = (__bf16)a0.x; wf[1]  = (__bf16)a0.y; wf[2]  = (__bf16)a0.z; wf[3]  = (__bf16)a0.w;
        wf[4]  = (__bf16)a1.x; wf[5]  = (__bf16)a1.y; wf[6]  = (__bf16)a1.z; wf[7]  = (__bf16)a1.w;
        wf[8]  = (__bf16)a2.x; wf[9]  = (__bf16)a2.y; wf[10] = (__bf16)a2.z; wf[11] = (__bf16)a2.w;
        wf[12] = (__bf16)a3.x; wf[13] = (__bf16)a3.y; wf[14] = (__bf16)a3.z; wf[15] = (__bf16)a3.w;
        *(v16bf*)&s_stage[wr_row * 32 + wr_col] = wf;
      }
      __syncthreads();
    }

    const __bf16* buf = &s_stage[USE_TDM ? ((c & 1) * CHUNK_ELEMS) : 0];

    // A fragment, contiguous k order (k = 16*half + s), matching B.
    v16bf af;
    {
      const int kb = k0 + half * 16;
      float4 f0 = *(const float4*)(xrow + kb);
      float4 f1 = *(const float4*)(xrow + kb + 4);
      float4 f2 = *(const float4*)(xrow + kb + 8);
      float4 f3 = *(const float4*)(xrow + kb + 12);
      af[0]  = (__bf16)f0.x; af[1]  = (__bf16)f0.y; af[2]  = (__bf16)f0.z; af[3]  = (__bf16)f0.w;
      af[4]  = (__bf16)f1.x; af[5]  = (__bf16)f1.y; af[6]  = (__bf16)f1.z; af[7]  = (__bf16)f1.w;
      af[8]  = (__bf16)f2.x; af[9]  = (__bf16)f2.y; af[10] = (__bf16)f2.z; af[11] = (__bf16)f2.w;
      af[12] = (__bf16)f3.x; af[13] = (__bf16)f3.y; af[14] = (__bf16)f3.z; af[15] = (__bf16)f3.w;
    }
    if (k0 + 32 < H_DIM) __builtin_prefetch(xrow + k0 + 32 + half * 8, 0, 0);

    #pragma unroll
    for (int t = 0; t < 8; ++t) {
      v16bf bf = *(const v16bf*)&buf[(t * 16 + nloc) * 32 + half * 16];
      acc[t] = __builtin_amdgcn_wmma_f32_16x16x32_bf16(
          false, af, false, bf, (short)0, acc[t], false, false);
    }

    if (USE_TDM) __syncthreads();   // all reads of buf done before TDM overwrite
  }

  // --- dump gate logits (tiles 0..3) per C layout: row m = half*8 + r ---
  {
    float* og = &s_out[wave * 1024];
    #pragma unroll
    for (int t = 0; t < 4; ++t)
      #pragma unroll
      for (int r = 0; r < 8; ++r)
        og[(half * 8 + r) * 64 + t * 16 + nloc] = acc[t][r];
  }

  // --- complexity from cp tiles (4..7) ---
  {
    float p8[8];
    #pragma unroll
    for (int r = 0; r < 8; ++r) p8[r] = 0.f;
    #pragma unroll
    for (int t = 4; t < 8; ++t) {
      const int   n  = (t - 4) * 16 + nloc;
      const float wt = s_w2[n];
      const float b1 = s_b1[n];
      #pragma unroll
      for (int r = 0; r < 8; ++r) {
        float hv = acc[t][r] + b1;
        p8[r] += fmaxf(hv, 0.f) * wt;
      }
    }
    #pragma unroll
    for (int msk = 1; msk <= 8; msk <<= 1)
      #pragma unroll
      for (int r = 0; r < 8; ++r)
        p8[r] += __shfl_xor(p8[r], msk, 32);
    if (nloc == 0) {
      const float b2 = cp_b2[0];
      float cs = 0.f;
      #pragma unroll
      for (int r = 0; r < 8; ++r) cs += sigmoidf_(p8[r] + b2);
      atomicAdd(&s_comp, cs);
    }
  }
  __syncthreads();

  // --- per-row softmax over 64 experts, top-2, aux-loss partials ---
  if (tid < 128) {
    const int wr = tid >> 4, m = tid & 15;
    const float* rowv = &s_out[wr * 1024 + m * 64];
    const int gm = mblk + wr * 16 + m;

    float v1 = -3.0e38f, v2 = -3.0e38f;
    int   i1 = 0, i2 = 0;
    for (int e = 0; e < 64; ++e) {
      float a = rowv[e] + s_eb[e] - s_fl[e];
      if (a > v1)      { v2 = v1; i2 = i1; v1 = a; i1 = e; }
      else if (a > v2) { v2 = a;  i2 = e; }
    }
    const float mx = v1;
    float sum = 0.f;
    for (int e = 0; e < 64; ++e)
      sum += __expf(rowv[e] + s_eb[e] - s_fl[e] - mx);
    const float inv = 1.f / sum;
    for (int e = 0; e < 64; ++e)
      atomicAdd(&s_P[e], __expf(rowv[e] + s_eb[e] - s_fl[e] - mx) * inv);

    const float s1 = __expf(v1 - mx) * inv;
    const float s2 = __expf(v2 - mx) * inv;
    const float dn = 1.f / (s1 + s2);
    out_scores[(size_t)gm * 2 + 0] = s1 * dn;
    out_scores[(size_t)gm * 2 + 1] = s2 * dn;
    out_idx[(size_t)gm * 2 + 0] = (float)i1;
    out_idx[(size_t)gm * 2 + 1] = (float)i2;
    atomicAdd(&s_cnt[i1], 1.f);
    atomicAdd(&s_cnt[i2], 1.f);
  }
  __syncthreads();

  if (tid < 64) {
    atomicAdd(&Pg[tid],   s_P[tid]);
    atomicAdd(&cntg[tid], s_cnt[tid]);
  }
  if (tid == 0) atomicAdd(compg, s_comp);
}

// ---------------------------------------------------------------------------
// Kernel 4: finalize aux_loss and capacity
// ---------------------------------------------------------------------------
__global__ __launch_bounds__(64) void moe_finalize(
    const float* __restrict__ Pg, const float* __restrict__ cntg,
    const float* __restrict__ compg,
    float* __restrict__ aux_out, float* __restrict__ cap_out, int N)
{
  __shared__ float red[64];
  const int e = threadIdx.x;
  const float f = cntg[e] / (float)(N * TOPK);
  const float P = Pg[e] / (float)N;
  red[e] = f * P;
  __syncthreads();
  if (e == 0) {
    float s = 0.f;
    for (int i = 0; i < 64; ++i) s += red[i];
    aux_out[0] = 64.f * s;
    float cap = 0.5f + 1.5f * (compg[0] / (float)N);
    cap = fminf(fmaxf(cap, 0.5f), 2.f);
    cap_out[0] = cap;
  }
}

// ---------------------------------------------------------------------------
extern "C" void kernel_launch(void* const* d_in, const int* in_sizes, int n_in,
                              void* d_out, int out_size, void* d_ws, size_t ws_size,
                              hipStream_t stream) {
  const float* x    = (const float*)d_in[0];
  const float* gw   = (const float*)d_in[1];
  const float* eb   = (const float*)d_in[2];
  const float* w1   = (const float*)d_in[3];
  const float* b1   = (const float*)d_in[4];
  const float* w2   = (const float*)d_in[5];
  const float* b2   = (const float*)d_in[6];
  const float* lb   = (const float*)d_in[7];
  const float* wih0 = (const float*)d_in[8];
  const float* whh0 = (const float*)d_in[9];
  const float* bih0 = (const float*)d_in[10];
  const float* bhh0 = (const float*)d_in[11];
  const float* wih1 = (const float*)d_in[12];
  const float* whh1 = (const float*)d_in[13];
  const float* bih1 = (const float*)d_in[14];
  const float* bhh1 = (const float*)d_in[15];
  const float* hw   = (const float*)d_in[16];
  const float* hb   = (const float*)d_in[17];

  const int N = in_sizes[0] / H_DIM;   // 16384 tokens

  float*  ws    = (float*)d_ws;
  float*  fl    = ws;         // [0:64]
  float*  Pg    = ws + 64;    // [64:128]
  float*  cntg  = ws + 128;   // [128:192]
  float*  compg = ws + 192;   // [192]
  __bf16* wbf   = (__bf16*)((char*)d_ws + 4096);   // 1 MB bf16 weight copy

  const size_t ws_needed = 4096 + (size_t)128 * H_DIM * 2;
  const bool use_tdm = (ws_size >= ws_needed);

  float* out     = (float*)d_out;
  float* out_sc  = out;
  float* out_idx = out + (size_t)N * 2;
  float* out_aux = out + (size_t)N * 4;
  float* out_cap = out_aux + 1;
  float* out_fl  = out_aux + 2;

  moe_zero_ws<<<1, 256, 0, stream>>>(ws);
  moe_lstm_pred<<<1, 256, 0, stream>>>(lb, wih0, whh0, bih0, bhh0,
                                       wih1, whh1, bih1, bhh1, hw, hb,
                                       fl, out_fl);
  if (use_tdm) {
    moe_convert_w<<<512, 256, 0, stream>>>(gw, w1, wbf);
    moe_gate_main<true><<<N / ROWS_PER_BLK, 256, 0, stream>>>(
        x, gw, eb, w1, b1, w2, b2, wbf, fl, Pg, cntg, compg, out_sc, out_idx, N);
  } else {
    moe_gate_main<false><<<N / ROWS_PER_BLK, 256, 0, stream>>>(
        x, gw, eb, w1, b1, w2, b2, nullptr, fl, Pg, cntg, compg, out_sc, out_idx, N);
  }
  moe_finalize<<<1, 64, 0, stream>>>(Pg, cntg, compg, out_aux, out_cap, N);
}
